// GraphFingerprintsModel_18726057410984
// MI455X (gfx1250) — matrix operand order repaired and checked by reading, not compiled, not run on hardware
//
#include <hip/hip_runtime.h>
#include <hip/hip_bf16.h>
#include <stdint.h>

#define GD 256
#define NN 32768
#define EE 131072
#define BB 256
#define NODEF 39
#define EDGEF 10
#define SOLVN 1024
#define EXTRAN 1168

typedef __attribute__((ext_vector_type(16))) __bf16 v16bf;
typedef __attribute__((ext_vector_type(8)))  float  v8f;
typedef __attribute__((ext_vector_type(4)))  unsigned int u32x4;

union ABfrag { u32x4 u[2]; v16bf v; };

__device__ __forceinline__ unsigned short f2bf(float f) {
    unsigned u = __float_as_uint(f);
    u += 0x7FFFu + ((u >> 16) & 1u);           // round-to-nearest-even
    return (unsigned short)(u >> 16);
}
__device__ __forceinline__ float bf2f(unsigned short h) {
    return __uint_as_float(((unsigned)h) << 16);
}
__device__ __forceinline__ float leakyf(float x) { return x > 0.f ? x : 0.01f * x; }
__device__ __forceinline__ unsigned fenc(float f) {
    unsigned u = __float_as_uint(f);
    return (u & 0x80000000u) ? ~u : (u | 0x80000000u);
}
__device__ __forceinline__ float fdec(unsigned u) {
    unsigned b = (u & 0x80000000u) ? (u & 0x7FFFFFFFu) : ~u;
    return __uint_as_float(b);
}
__device__ __forceinline__ float wredsum(float v) {
    #pragma unroll
    for (int o = 16; o; o >>= 1) v += __shfl_xor(v, o, 32);
    return v;
}

// ---------------- elementwise converts ----------------
__global__ void k_cvt(const float* __restrict__ s, unsigned short* __restrict__ d, int n) {
    int i = blockIdx.x * blockDim.x + threadIdx.x;
    if (i < n) d[i] = f2bf(s[i]);
}
__global__ void k_relu_cvt(const float* __restrict__ s, unsigned short* __restrict__ d, int n) {
    int i = blockIdx.x * blockDim.x + threadIdx.x;
    if (i < n) d[i] = f2bf(fmaxf(s[i], 0.f));
}
__global__ void k_elu_cvt(const float* __restrict__ s, unsigned short* __restrict__ d, int n) {
    int i = blockIdx.x * blockDim.x + threadIdx.x;
    if (i < n) { float x = s[i]; d[i] = f2bf(x > 0.f ? x : (__expf(x) - 1.f)); }
}
__global__ void k_cvt_rows(const float* __restrict__ s, int stride, int off,
                           unsigned short* __restrict__ d, int rows, int cols) {
    int i = blockIdx.x * blockDim.x + threadIdx.x;
    if (i >= rows * cols) return;
    int r = i / cols, c = i - r * cols;
    d[i] = f2bf(s[(size_t)r * stride + off + c]);
}

// ---------------- thin-K projections (K=39 / K=49) ----------------
__global__ void k_node_proj(const float* __restrict__ nf, const float* __restrict__ W,
                            const float* __restrict__ b, float* __restrict__ hv,
                            unsigned short* __restrict__ hv_bf) {
    __shared__ float wsh[GD * NODEF];
    __shared__ float xr[48];
    int tid = threadIdx.x;
    for (int i = tid; i < GD * NODEF; i += 256) wsh[i] = W[i];
    __syncthreads();
    int row0 = blockIdx.x * 16;
    for (int r = 0; r < 16; ++r) {
        int row = row0 + r;
        if (tid < NODEF) xr[tid] = nf[(size_t)row * NODEF + tid];
        __syncthreads();
        float acc = b[tid];
        #pragma unroll
        for (int k = 0; k < NODEF; ++k) acc += xr[k] * wsh[tid * NODEF + k];
        acc = leakyf(acc);
        size_t o = (size_t)row * GD + tid;
        hv[o] = acc; hv_bf[o] = f2bf(acc);
        __syncthreads();
    }
}

__global__ void k_edge_proj(const float* __restrict__ nf, const float* __restrict__ ef,
                            const int* __restrict__ src, const float* __restrict__ W,
                            const float* __restrict__ b, unsigned short* __restrict__ he) {
    __shared__ float wsh[GD * 49];
    __shared__ float xr[52];
    int tid = threadIdx.x;
    for (int i = tid; i < GD * 49; i += 256) wsh[i] = W[i];
    __syncthreads();
    int row0 = blockIdx.x * 16;
    for (int r = 0; r < 16; ++r) {
        int row = row0 + r;
        if (tid < NODEF) xr[tid] = nf[(size_t)src[row] * NODEF + tid];
        else if (tid < NODEF + EDGEF) xr[tid] = ef[(size_t)row * EDGEF + (tid - NODEF)];
        __syncthreads();
        float acc = b[tid];
        #pragma unroll
        for (int k = 0; k < 49; ++k) acc += xr[k] * wsh[tid * 49 + k];
        he[(size_t)row * GD + tid] = f2bf(leakyf(acc));
        __syncthreads();
    }
}

// ---- generic bf16 WMMA GEMM, 16x64 tile per wave: out = A[MxK] @ W[NoutxK]^T + bias ----
// K is compile-time -> fully-unrolled K loop, loads scheduled across WMMAs (partial waits),
// branch-free prefetch. One A fragment feeds 4 independent WMMA chains.
template <int K>
__global__ void k_gemm_wmma(const unsigned short* __restrict__ A,
                            const unsigned short* __restrict__ W,
                            const float* __restrict__ bias,
                            float* __restrict__ out, unsigned short* __restrict__ obf,
                            int M, int Nout, int act) {
    int wave = (blockIdx.x * blockDim.x + threadIdx.x) >> 5;
    int ntile = Nout >> 6;                      // 64 output columns per wave
    int mt = wave / ntile;
    int nt = wave - mt * ntile;
    if (mt >= (M >> 4)) return;
    int lane = threadIdx.x & 31;
    int half = lane >> 4;
    const unsigned short* arow = A + (size_t)(mt * 16 + (lane & 15)) * K;
    const unsigned short* brow = W + (size_t)(nt * 64 + (lane & 15)) * K + half * 16;
    size_t bstep = (size_t)16 * K;              // 16 output columns
    v8f ac0 = {}, ac1 = {}, ac2 = {}, ac3 = {};
    #pragma unroll
    for (int k0 = 0; k0 < K; k0 += 32) {
        if (k0 + 64 < K) __builtin_prefetch(arow + k0 + 64, 0, 3);
        ABfrag af, b0, b1, b2, b3;
        af.u[0] = *(const u32x4*)(arow + k0 + half * 16);
        af.u[1] = *(const u32x4*)(arow + k0 + half * 16 + 8);
        const unsigned short* bp = brow + k0;
        b0.u[0] = *(const u32x4*)(bp);                b0.u[1] = *(const u32x4*)(bp + 8);
        b1.u[0] = *(const u32x4*)(bp + bstep);        b1.u[1] = *(const u32x4*)(bp + bstep + 8);
        b2.u[0] = *(const u32x4*)(bp + 2 * bstep);    b2.u[1] = *(const u32x4*)(bp + 2 * bstep + 8);
        b3.u[0] = *(const u32x4*)(bp + 3 * bstep);    b3.u[1] = *(const u32x4*)(bp + 3 * bstep + 8);
        ac0 = __builtin_amdgcn_wmma_f32_16x16x32_bf16(false, af.v, false, b0.v, (short)0, ac0, false, false);
        ac1 = __builtin_amdgcn_wmma_f32_16x16x32_bf16(false, af.v, false, b1.v, (short)0, ac1, false, false);
        ac2 = __builtin_amdgcn_wmma_f32_16x16x32_bf16(false, af.v, false, b2.v, (short)0, ac2, false, false);
        ac3 = __builtin_amdgcn_wmma_f32_16x16x32_bf16(false, af.v, false, b3.v, (short)0, ac3, false, false);
    }
    v8f accs[4] = { ac0, ac1, ac2, ac3 };
    int rbase = mt * 16 + half * 8;
    #pragma unroll
    for (int c = 0; c < 4; ++c) {
        int col = nt * 64 + c * 16 + (lane & 15);
        float bc = bias[col];
        #pragma unroll
        for (int j = 0; j < 8; ++j) {
            float v = accs[c][j] + bc;
            if (act == 1) v = fmaxf(v, 0.f);
            else if (act == 2) v = leakyf(v);
            size_t idx = (size_t)(rbase + j) * Nout + col;
            out[idx] = v;
            if (obf) obf[idx] = f2bf(v);
        }
    }
}

// -- W_et GEMM (16x64 tile) fused with softmax-scale + segment-sum epilogue (fp32 L2 atomics) --
__global__ void k_msg_wmma(const unsigned short* __restrict__ he,
                           const unsigned short* __restrict__ Wet,
                           const float* __restrict__ bias,
                           const float* __restrict__ a,
                           const int* __restrict__ dst,
                           float* __restrict__ ctx) {
    int wave = (blockIdx.x * blockDim.x + threadIdx.x) >> 5;
    const int ntile = GD >> 6;                  // 4
    int mt = wave / ntile;
    int nt = wave - mt * ntile;
    if (mt >= (EE >> 4)) return;
    int lane = threadIdx.x & 31;
    int half = lane >> 4;
    const unsigned short* arow = he + (size_t)(mt * 16 + (lane & 15)) * GD;
    const unsigned short* brow = Wet + (size_t)(nt * 64 + (lane & 15)) * GD + half * 16;
    const size_t bstep = (size_t)16 * GD;
    v8f ac0 = {}, ac1 = {}, ac2 = {}, ac3 = {};
    #pragma unroll
    for (int k0 = 0; k0 < GD; k0 += 32) {
        ABfrag af, b0, b1, b2, b3;
        af.u[0] = *(const u32x4*)(arow + k0 + half * 16);
        af.u[1] = *(const u32x4*)(arow + k0 + half * 16 + 8);
        const unsigned short* bp = brow + k0;
        b0.u[0] = *(const u32x4*)(bp);                b0.u[1] = *(const u32x4*)(bp + 8);
        b1.u[0] = *(const u32x4*)(bp + bstep);        b1.u[1] = *(const u32x4*)(bp + bstep + 8);
        b2.u[0] = *(const u32x4*)(bp + 2 * bstep);    b2.u[1] = *(const u32x4*)(bp + 2 * bstep + 8);
        b3.u[0] = *(const u32x4*)(bp + 3 * bstep);    b3.u[1] = *(const u32x4*)(bp + 3 * bstep + 8);
        ac0 = __builtin_amdgcn_wmma_f32_16x16x32_bf16(false, af.v, false, b0.v, (short)0, ac0, false, false);
        ac1 = __builtin_amdgcn_wmma_f32_16x16x32_bf16(false, af.v, false, b1.v, (short)0, ac1, false, false);
        ac2 = __builtin_amdgcn_wmma_f32_16x16x32_bf16(false, af.v, false, b2.v, (short)0, ac2, false, false);
        ac3 = __builtin_amdgcn_wmma_f32_16x16x32_bf16(false, af.v, false, b3.v, (short)0, ac3, false, false);
    }
    v8f accs[4] = { ac0, ac1, ac2, ac3 };
    int ebase = mt * 16 + half * 8;
    #pragma unroll
    for (int c = 0; c < 4; ++c) {
        int col = nt * 64 + c * 16 + (lane & 15);
        float bc = bias[col];
        #pragma unroll
        for (int j = 0; j < 8; ++j) {
            int e = ebase + j;
            float val = a[e] * (accs[c][j] + bc);
            atomicAdd(ctx + (size_t)dst[e] * GD + col, val);
        }
    }
}

// ---------------- attention logits (one wave per row, 512-long dots) ----------------
__global__ void k_edge_logits1(const float* __restrict__ hv, const unsigned short* __restrict__ he,
                               const int* __restrict__ dst, const float* __restrict__ Wv,
                               const float* __restrict__ b0, float* __restrict__ out) {
    int w = (blockIdx.x * blockDim.x + threadIdx.x) >> 5;
    if (w >= EE) return;
    int lane = threadIdx.x & 31;
    const float* hrow = hv + (size_t)dst[w] * GD;
    const unsigned short* erow = he + (size_t)w * GD;
    float s = 0.f;
    for (int c = lane; c < GD; c += 32) s += hrow[c] * Wv[c] + bf2f(erow[c]) * Wv[GD + c];
    s = wredsum(s);
    if (lane == 0) out[w] = leakyf(s + b0[0]);
}
__global__ void k_edge_logits2(const float* __restrict__ h, const int* __restrict__ src,
                               const int* __restrict__ dst, const float* __restrict__ Wv,
                               const float* __restrict__ b0, float* __restrict__ out) {
    int w = (blockIdx.x * blockDim.x + threadIdx.x) >> 5;
    if (w >= EE) return;
    int lane = threadIdx.x & 31;
    const float* drow = h + (size_t)dst[w] * GD;
    const float* srow = h + (size_t)src[w] * GD;
    float s = 0.f;
    for (int c = lane; c < GD; c += 32) s += drow[c] * Wv[c] + srow[c] * Wv[GD + c];
    s = wredsum(s);
    if (lane == 0) out[w] = leakyf(s + b0[0]);
}
__global__ void k_node_logits(const float* __restrict__ g, const int* __restrict__ ng,
                              const float* __restrict__ h, const float* __restrict__ Wv,
                              const float* __restrict__ b0, float* __restrict__ out) {
    int w = (blockIdx.x * blockDim.x + threadIdx.x) >> 5;
    if (w >= NN) return;
    int lane = threadIdx.x & 31;
    const float* grow = g + (size_t)ng[w] * GD;
    const float* hrow = h + (size_t)w * GD;
    float s = 0.f;
    for (int c = lane; c < GD; c += 32) s += fmaxf(grow[c], 0.f) * Wv[c] + hrow[c] * Wv[GD + c];
    s = wredsum(s);
    if (lane == 0) out[w] = leakyf(s + b0[0]);
}

// ---------------- segment softmax (3 passes; monotonic-uint atomicMax) ----------------
__global__ void k_seg_amax(const float* __restrict__ v, const int* __restrict__ seg,
                           unsigned* __restrict__ m, int n) {
    int i = blockIdx.x * blockDim.x + threadIdx.x;
    if (i < n) atomicMax(m + seg[i], fenc(v[i]));
}
__global__ void k_seg_expsum(float* __restrict__ v, const int* __restrict__ seg,
                             const unsigned* __restrict__ m, float* __restrict__ s, int n) {
    int i = blockIdx.x * blockDim.x + threadIdx.x;
    if (i >= n) return;
    int sg = seg[i];
    float e = __expf(v[i] - fdec(m[sg]));
    v[i] = e;
    atomicAdd(s + sg, e);
}
__global__ void k_seg_norm(float* __restrict__ v, const int* __restrict__ seg,
                           const float* __restrict__ s, int n) {
    int i = blockIdx.x * blockDim.x + threadIdx.x;
    if (i < n) v[i] = v[i] / s[seg[i]];
}

// ---------------- gathers / scatters (wave per row) ----------------
__global__ void k_scatter_edge(const float* __restrict__ hv2, const int* __restrict__ src,
                               const int* __restrict__ dst, const float* __restrict__ a,
                               float* __restrict__ ctx) {
    int w = (blockIdx.x * blockDim.x + threadIdx.x) >> 5;
    if (w >= EE) return;
    int lane = threadIdx.x & 31;
    float av = a[w];
    const float* row = hv2 + (size_t)src[w] * GD;
    float* orow = ctx + (size_t)dst[w] * GD;
    for (int c = lane; c < GD; c += 32) atomicAdd(orow + c, av * row[c]);
}
__global__ void k_gfeats(const float* __restrict__ h, const int* __restrict__ ng,
                         float* __restrict__ g) {
    int w = (blockIdx.x * blockDim.x + threadIdx.x) >> 5;
    if (w >= NN) return;
    int lane = threadIdx.x & 31;
    float* grow = g + (size_t)ng[w] * GD;
    const float* hrow = h + (size_t)w * GD;
    for (int c = lane; c < GD; c += 32) atomicAdd(grow + c, hrow[c]);
}
__global__ void k_scatter_node(const float* __restrict__ hv, const int* __restrict__ ng,
                               const float* __restrict__ a, float* __restrict__ gr) {
    int w = (blockIdx.x * blockDim.x + threadIdx.x) >> 5;
    if (w >= NN) return;
    int lane = threadIdx.x & 31;
    float av = a[w];
    float* grow = gr + (size_t)ng[w] * GD;
    const float* hrow = hv + (size_t)w * GD;
    for (int c = lane; c < GD; c += 32) atomicAdd(grow + c, av * hrow[c]);
}

// ---------------- GRU gate math (biases pre-folded into gi/gh GEMMs) ----------------
__global__ void k_gru(const float* __restrict__ gi, const float* __restrict__ gh,
                      const float* __restrict__ hprev, float* __restrict__ hout,
                      unsigned short* __restrict__ hbf, int M, int relu_out) {
    int idx = blockIdx.x * blockDim.x + threadIdx.x;
    if (idx >= M * GD) return;
    int n = idx / GD, c = idx - n * GD;
    size_t b3 = (size_t)n * (3 * GD);
    float ir = gi[b3 + c], iz = gi[b3 + GD + c], in_ = gi[b3 + 2 * GD + c];
    float hr = gh[b3 + c], hz = gh[b3 + GD + c], hn  = gh[b3 + 2 * GD + c];
    float r = 1.f / (1.f + __expf(-(ir + hr)));
    float z = 1.f / (1.f + __expf(-(iz + hz)));
    float nn = tanhf(in_ + r * hn);
    float h = (1.f - z) * nn + z * hprev[idx];
    if (relu_out) h = fmaxf(h, 0.f);
    hout[idx] = h;
    if (hbf) hbf[idx] = f2bf(h);
}

// ---------------- conv1d(k=3,pad=1) + online softmax attention + maxpool ----------------
__global__ void k_conv(const float* __restrict__ fp,
                       const float* __restrict__ wf, const float* __restrict__ bfeat,
                       const float* __restrict__ wa, const float* __restrict__ ba,
                       float* __restrict__ attn_out, float* __restrict__ pooled) {
    int w = (blockIdx.x * blockDim.x + threadIdx.x) >> 5;
    if (w >= BB * GD) return;
    int b = w / GD, g = w - b * GD;
    int lane = threadIdx.x & 31;
    const float* x = fp + (size_t)b * (SOLVN + EXTRAN) + SOLVN;
    float wf0 = wf[g * 3], wf1 = wf[g * 3 + 1], wf2 = wf[g * 3 + 2], fb = bfeat[g];
    float wa0 = wa[g * 3], wa1 = wa[g * 3 + 1], wa2 = wa[g * 3 + 2], ab = ba[g];
    float m = -3.0e38f, s = 0.f, ws = 0.f, pool = -3.0e38f;
    for (int l = lane; l < EXTRAN; l += 32) {
        float xm = (l > 0) ? x[l - 1] : 0.f;
        float x0 = x[l];
        float xp = (l < EXTRAN - 1) ? x[l + 1] : 0.f;
        float feat = wf0 * xm + wf1 * x0 + wf2 * xp + fb;
        float attn = wa0 * xm + wa1 * x0 + wa2 * xp + ab;
        pool = fmaxf(pool, feat);
        if (attn > m) {
            float c = __expf(m - attn);
            s = s * c + 1.f; ws = ws * c + feat; m = attn;
        } else {
            float e = __expf(attn - m);
            s += e; ws += e * feat;
        }
    }
    #pragma unroll
    for (int o = 16; o; o >>= 1) {
        float m2 = __shfl_xor(m, o, 32), s2 = __shfl_xor(s, o, 32);
        float w2 = __shfl_xor(ws, o, 32), p2 = __shfl_xor(pool, o, 32);
        float M2 = fmaxf(m, m2);
        float c1 = __expf(m - M2), c2 = __expf(m2 - M2);
        s = s * c1 + s2 * c2; ws = ws * c1 + w2 * c2; m = M2; pool = fmaxf(pool, p2);
    }
    if (lane == 0) { attn_out[w] = ws / s; pooled[w] = pool; }
}

// ---------------- final MLP head: [B,1024] -> relu[B,128] -> [B,1] ----------------
__global__ void k_final(const float* __restrict__ g, const float* __restrict__ solv,
                        const float* __restrict__ attn_out, const float* __restrict__ pooled,
                        const float* __restrict__ W1, const float* __restrict__ b1,
                        const float* __restrict__ W2, const float* __restrict__ b2,
                        float* __restrict__ out) {
    __shared__ float comb[1024];
    __shared__ float red[128];
    int b = blockIdx.x, tid = threadIdx.x;
    for (int i = tid; i < GD; i += 128) {
        comb[i]          = g[(size_t)b * GD + i];
        comb[GD + i]     = solv[(size_t)b * GD + i];
        comb[2 * GD + i] = attn_out[(size_t)b * GD + i];
        comb[3 * GD + i] = pooled[(size_t)b * GD + i];
    }
    __syncthreads();
    float acc = b1[tid];
    const float* wrow = W1 + (size_t)tid * 1024;
    for (int k = 0; k < 1024; ++k) acc += comb[k] * wrow[k];
    red[tid] = fmaxf(acc, 0.f) * W2[tid];
    __syncthreads();
    for (int o = 64; o; o >>= 1) { if (tid < o) red[tid] += red[tid + o]; __syncthreads(); }
    if (tid == 0) out[b] = red[0] + b2[0];
}

// =======================================================================================
extern "C" void kernel_launch(void* const* d_in, const int* in_sizes, int n_in,
                              void* d_out, int out_size, void* d_ws, size_t ws_size,
                              hipStream_t stream) {
    (void)in_sizes; (void)n_in; (void)out_size; (void)ws_size;
    const float* node_feats = (const float*)d_in[0];
    const float* edge_feats = (const float*)d_in[1];
    const float* fingerprints = (const float*)d_in[2];
    const int* src = (const int*)d_in[3];
    const int* dst = (const int*)d_in[4];
    const int* node_graph = (const int*)d_in[5];
    const float* W_pn = (const float*)d_in[6];  const float* b_pn = (const float*)d_in[7];
    const float* W_pe1 = (const float*)d_in[8]; const float* b_pe1 = (const float*)d_in[9];
    const float* W_pe2 = (const float*)d_in[10];const float* b_pe2 = (const float*)d_in[11];
    const float* W_et = (const float*)d_in[12]; const float* b_et = (const float*)d_in[13];
    const float* g0Wih = (const float*)d_in[14];const float* g0Whh = (const float*)d_in[15];
    const float* g0bih = (const float*)d_in[16];const float* g0bhh = (const float*)d_in[17];
    const float* W_pe = (const float*)d_in[18]; const float* b_pe = (const float*)d_in[19];
    const float* W_pn2 = (const float*)d_in[20];const float* b_pn2 = (const float*)d_in[21];
    const float* g1Wih = (const float*)d_in[22];const float* g1Whh = (const float*)d_in[23];
    const float* g1bih = (const float*)d_in[24];const float* g1bhh = (const float*)d_in[25];
    const float* roWcl = (const float*)d_in[26];const float* robcl = (const float*)d_in[27];
    const float* roWpn = (const float*)d_in[28];const float* robpn = (const float*)d_in[29];
    const float* roWih = (const float*)d_in[30];const float* roWhh = (const float*)d_in[31];
    const float* robih = (const float*)d_in[32];const float* robhh = (const float*)d_in[33];
    const float* cfw = (const float*)d_in[34];  const float* cfb = (const float*)d_in[35];
    const float* caw = (const float*)d_in[36];  const float* cab = (const float*)d_in[37];
    const float* W_s1 = (const float*)d_in[38]; const float* b_s1 = (const float*)d_in[39];
    const float* W_s2 = (const float*)d_in[40]; const float* b_s2 = (const float*)d_in[41];
    const float* W_p1 = (const float*)d_in[42]; const float* b_p1 = (const float*)d_in[43];
    const float* W_p2 = (const float*)d_in[44]; const float* b_p2 = (const float*)d_in[45];

    // ---- workspace carve-up ----
    char* base = (char*)d_ws;
    size_t off = 0;
    auto alloc = [&](size_t bytes) -> void* {
        void* r = base + off; off = (off + bytes + 255) & ~(size_t)255; return r;
    };
    float* hv             = (float*)alloc((size_t)NN * GD * 4);   // hv_new -> hv2 -> readout hv
    unsigned short* hv_bf = (unsigned short*)alloc((size_t)NN * GD * 2); // hv_new bf16 -> h bf16
    unsigned short* he_bf = (unsigned short*)alloc((size_t)EE * GD * 2);
    float* av             = (float*)alloc((size_t)EE * 4);        // logits -> softmax weights
    unsigned* mEnc        = (unsigned*)alloc((size_t)NN * 4);
    float* sSum           = (float*)alloc((size_t)NN * 4);
    float* ctx            = (float*)alloc((size_t)NN * GD * 4);   // ctx / ctx2
    unsigned short* ctx_bf= (unsigned short*)alloc((size_t)NN * GD * 2);
    float* gi             = (float*)alloc((size_t)NN * 3 * GD * 4);
    float* gh             = (float*)alloc((size_t)NN * 3 * GD * 4);
    float* hbuf           = (float*)alloc((size_t)NN * GD * 4);
    float* gfe            = (float*)alloc((size_t)BB * GD * 4);
    unsigned short* g_bf  = (unsigned short*)alloc((size_t)BB * GD * 2);
    float* grepr          = (float*)alloc((size_t)BB * GD * 4);
    unsigned short* gx_bf = (unsigned short*)alloc((size_t)BB * GD * 2);
    unsigned short* solv_bf = (unsigned short*)alloc((size_t)BB * SOLVN * 2);
    float* shid           = (float*)alloc((size_t)BB * GD * 4);
    unsigned short* shid_bf = (unsigned short*)alloc((size_t)BB * GD * 2);
    float* sout           = (float*)alloc((size_t)BB * GD * 4);
    float* attn_out       = (float*)alloc((size_t)BB * GD * 4);
    float* pooled         = (float*)alloc((size_t)BB * GD * 4);
    unsigned short* Wet_bf  = (unsigned short*)alloc((size_t)GD * GD * 2);
    unsigned short* Wih0_bf = (unsigned short*)alloc((size_t)3 * GD * GD * 2);
    unsigned short* Whh0_bf = (unsigned short*)alloc((size_t)3 * GD * GD * 2);
    unsigned short* Wpn2_bf = (unsigned short*)alloc((size_t)GD * GD * 2);
    unsigned short* Wih1_bf = (unsigned short*)alloc((size_t)3 * GD * GD * 2);
    unsigned short* Whh1_bf = (unsigned short*)alloc((size_t)3 * GD * GD * 2);
    unsigned short* roWpn_bf= (unsigned short*)alloc((size_t)2 * GD * GD * 2);
    unsigned short* roWih_bf= (unsigned short*)alloc((size_t)2 * 3 * GD * GD * 2);
    unsigned short* roWhh_bf= (unsigned short*)alloc((size_t)2 * 3 * GD * GD * 2);
    unsigned short* Ws1_bf  = (unsigned short*)alloc((size_t)GD * SOLVN * 2);
    unsigned short* Ws2_bf  = (unsigned short*)alloc((size_t)GD * GD * 2);

    auto cvt = [&](const float* s, unsigned short* d, int n) {
        k_cvt<<<(n + 255) / 256, 256, 0, stream>>>(s, d, n);
    };
    auto gemm = [&](const unsigned short* A, const unsigned short* W, const float* bias,
                    float* o, unsigned short* obf, int M, int K, int Nout, int act) {
        int waves = (M / 16) * (Nout / 64);
        int blocks = (waves + 7) / 8;
        if (K == SOLVN)
            k_gemm_wmma<SOLVN><<<blocks, 256, 0, stream>>>(A, W, bias, o, obf, M, Nout, act);
        else
            k_gemm_wmma<GD><<<blocks, 256, 0, stream>>>(A, W, bias, o, obf, M, Nout, act);
    };
    auto seg_softmax = [&](float* logits, const int* seg, int n, int nseg) {
        hipMemsetAsync(mEnc, 0, (size_t)nseg * 4, stream);
        hipMemsetAsync(sSum, 0, (size_t)nseg * 4, stream);
        int blk = (n + 255) / 256;
        k_seg_amax<<<blk, 256, 0, stream>>>(logits, seg, mEnc, n);
        k_seg_expsum<<<blk, 256, 0, stream>>>(logits, seg, mEnc, sSum, n);
        k_seg_norm<<<blk, 256, 0, stream>>>(logits, seg, sSum, n);
    };

    // ---- weight bf16 conversions ----
    cvt(W_et, Wet_bf, GD * GD);
    cvt(g0Wih, Wih0_bf, 3 * GD * GD);  cvt(g0Whh, Whh0_bf, 3 * GD * GD);
    cvt(W_pn2, Wpn2_bf, GD * GD);
    cvt(g1Wih, Wih1_bf, 3 * GD * GD);  cvt(g1Whh, Whh1_bf, 3 * GD * GD);
    cvt(roWpn, roWpn_bf, 2 * GD * GD);
    cvt(roWih, roWih_bf, 2 * 3 * GD * GD);  cvt(roWhh, roWhh_bf, 2 * 3 * GD * GD);
    cvt(W_s1, Ws1_bf, GD * SOLVN);  cvt(W_s2, Ws2_bf, GD * GD);

    // ---- message pass 1 ----
    k_node_proj<<<NN / 16, 256, 0, stream>>>(node_feats, W_pn, b_pn, hv, hv_bf);
    k_edge_proj<<<EE / 16, 256, 0, stream>>>(node_feats, edge_feats, src, W_pe1, b_pe1, he_bf);
    k_edge_logits1<<<EE * 32 / 256, 256, 0, stream>>>(hv, he_bf, dst, W_pe2, b_pe2, av);
    seg_softmax(av, dst, EE, NN);
    hipMemsetAsync(ctx, 0, (size_t)NN * GD * 4, stream);
    k_msg_wmma<<<(EE / 16) * (GD / 64) / 8, 256, 0, stream>>>(he_bf, Wet_bf, b_et, av, dst, ctx);
    k_elu_cvt<<<(NN * GD + 255) / 256, 256, 0, stream>>>(ctx, ctx_bf, NN * GD);
    gemm(ctx_bf, Wih0_bf, g0bih, gi, nullptr, NN, GD, 3 * GD, 0);
    gemm(hv_bf,  Whh0_bf, g0bhh, gh, nullptr, NN, GD, 3 * GD, 0);
    k_gru<<<(NN * GD + 255) / 256, 256, 0, stream>>>(gi, gh, hv, hbuf, hv_bf, NN, 1);

    // ---- message pass 2 ----
    k_edge_logits2<<<EE * 32 / 256, 256, 0, stream>>>(hbuf, src, dst, W_pe, b_pe, av);
    seg_softmax(av, dst, EE, NN);
    gemm(hv_bf, Wpn2_bf, b_pn2, hv, nullptr, NN, GD, GD, 0);      // hv2 = h @ W_pn2^T + b
    hipMemsetAsync(ctx, 0, (size_t)NN * GD * 4, stream);
    k_scatter_edge<<<EE * 32 / 256, 256, 0, stream>>>(hv, src, dst, av, ctx);
    k_elu_cvt<<<(NN * GD + 255) / 256, 256, 0, stream>>>(ctx, ctx_bf, NN * GD);
    gemm(ctx_bf, Wih1_bf, g1bih, gi, nullptr, NN, GD, 3 * GD, 0);
    gemm(hv_bf,  Whh1_bf, g1bhh, gh, nullptr, NN, GD, 3 * GD, 0);
    k_gru<<<(NN * GD + 255) / 256, 256, 0, stream>>>(gi, gh, hbuf, hbuf, hv_bf, NN, 1);

    // ---- graph readout (T=2) ----
    hipMemsetAsync(gfe, 0, (size_t)BB * GD * 4, stream);
    k_gfeats<<<NN * 32 / 256, 256, 0, stream>>>(hbuf, node_graph, gfe);
    for (int t = 0; t < 2; ++t) {
        k_node_logits<<<NN * 32 / 256, 256, 0, stream>>>(gfe, node_graph, hbuf,
                                                         roWcl + t * 2 * GD, robcl + t, av);
        seg_softmax(av, node_graph, NN, BB);
        gemm(hv_bf, roWpn_bf + (size_t)t * GD * GD, robpn + t * GD, hv, nullptr, NN, GD, GD, 0);
        hipMemsetAsync(grepr, 0, (size_t)BB * GD * 4, stream);
        k_scatter_node<<<NN * 32 / 256, 256, 0, stream>>>(hv, node_graph, av, grepr);
        k_relu_cvt<<<(BB * GD + 255) / 256, 256, 0, stream>>>(grepr, gx_bf, BB * GD);
        k_cvt<<<(BB * GD + 255) / 256, 256, 0, stream>>>(gfe, g_bf, BB * GD);
        gemm(gx_bf, roWih_bf + (size_t)t * 3 * GD * GD, robih + t * 3 * GD, gi, nullptr, BB, GD, 3 * GD, 0);
        gemm(g_bf,  roWhh_bf + (size_t)t * 3 * GD * GD, robhh + t * 3 * GD, gh, nullptr, BB, GD, 3 * GD, 0);
        k_gru<<<(BB * GD + 255) / 256, 256, 0, stream>>>(gi, gh, gfe, gfe, nullptr, BB, 0);
    }

    // ---- fingerprint branch ----
    k_cvt_rows<<<(BB * SOLVN + 255) / 256, 256, 0, stream>>>(fingerprints, SOLVN + EXTRAN, 0,
                                                             solv_bf, BB, SOLVN);
    gemm(solv_bf, Ws1_bf, b_s1, shid, shid_bf, BB, SOLVN, GD, 1);
    gemm(shid_bf, Ws2_bf, b_s2, sout, nullptr, BB, GD, GD, 0);
    k_conv<<<BB * GD * 32 / 256, 256, 0, stream>>>(fingerprints, cfw, cfb, caw, cab,
                                                   attn_out, pooled);

    // ---- head ----
    k_final<<<BB, 128, 0, stream>>>(gfe, sout, attn_out, pooled, W_p1, b_p1, W_p2, b_p2,
                                    (float*)d_out);
}
